// LRNet_65343632441817
// MI455X (gfx1250) — matrix operand order repaired and checked by reading, not compiled, not run on hardware
//
#include <hip/hip_runtime.h>
#include <hip/hip_bf16.h>

// ---------------- problem constants (match reference) ----------------
#define NN     8192
#define NFEAT  500
#define NHID   256
#define NCLS   16
#define KK     64
#define EE     262144
#define EPSC   0.1f
#define BETAC  1.0f
#define ALPHAC 0.1f
#define LAMDAC 0.5f

// ---------------- types ----------------
typedef __attribute__((ext_vector_type(16))) __bf16 v16bf;
typedef __attribute__((ext_vector_type(8)))  float  v8f;
typedef unsigned int u32x4 __attribute__((ext_vector_type(4)));
typedef int          i32x8 __attribute__((ext_vector_type(8)));
typedef int          i32x4 __attribute__((ext_vector_type(4)));

// native cast -> hardware bf16 convert (RNE), avoids manual v_add3/v_bfe chain
static __device__ __forceinline__ __bf16 f2bf(float f) { return (__bf16)f; }

#if __has_builtin(__builtin_amdgcn_tensor_load_to_lds)
#define HAS_TDM 1
#else
#define HAS_TDM 0
#endif

#if HAS_TDM
// TDM load of a [rows=32] x [cols=tileW] f32 tile (row stride = ldb elems)
// into LDS at ldsOff. tensor_dim1 = remaining K rows so TDM's OOB-zero
// covers the ragged-K tail. D# layout per CDNA5 ISA 8.3/8.4; 6-arg builtin.
static __device__ __forceinline__ void tdm_load_tile(const float* gptr, unsigned ldsOff,
                                                     unsigned dim0Extent, long kRows,
                                                     unsigned ldbElems, unsigned tileW)
{
    unsigned long long ga = (unsigned long long)(uintptr_t)gptr;
    unsigned td0 = dim0Extent;                       // tensor_dim0 (elems)
    unsigned td1 = (unsigned)(kRows > 0 ? kRows : 0);// tensor_dim1 (elems)
    u32x4 g0;
    g0[0] = 1u;                                      // count=1, user descriptor
    g0[1] = ldsOff;                                  // lds_addr (bytes)
    g0[2] = (unsigned)(ga & 0xFFFFFFFFu);            // global_addr[31:0]
    g0[3] = (unsigned)((ga >> 32) & 0x01FFFFFFu)     // global_addr[56:32]
            | (2u << 30);                            // type=2 ("image")
    i32x8 g1;
    g1[0] = (int)(2u << 16);                         // data_size=2 -> 4 bytes
    g1[1] = (int)((td0 & 0xFFFFu) << 16);            // tensor_dim0[15:0] @ bits63:48
    g1[2] = (int)((td0 >> 16) | ((td1 & 0xFFFFu) << 16)); // td0[31:16], td1[15:0]
    g1[3] = (int)((td1 >> 16) | (tileW << 16));      // td1[31:16], tile_dim0=tileW
    g1[4] = (int)32u;                                // tile_dim1=32, tile_dim2=0
    g1[5] = (int)ldbElems;                           // tensor_dim0_stride[31:0]
    g1[6] = 0;                                       // stride0[47:32]=0, stride1 lo=0
    g1[7] = 0;
    i32x4 z4 = {0, 0, 0, 0};
    i32x8 z8 = {0, 0, 0, 0, 0, 0, 0, 0};
    __builtin_amdgcn_tensor_load_to_lds(g0, g1, z4, z4, z8, 0);
}
#endif

// ---- A-fragment loader: 16 bf16 per lane (k = 16*hf + s, contiguous; B uses the same order) ----
template<bool TA, bool GUARD>
static __device__ __forceinline__ v16bf load_a_frag(const float* __restrict__ A, int lda,
                                                    int arow, int k0, int hf, int Kd)
{
    v16bf a;
    if (!TA && !GUARD) {
        // one contiguous 16-float chunk -> 4x b128 loads
        const float* ap = A + (long)arow * lda + k0 + hf * 16;
        float4 x0 = *(const float4*)(ap);
        float4 x1 = *(const float4*)(ap + 4);
        float4 x2 = *(const float4*)(ap + 8);
        float4 x3 = *(const float4*)(ap + 12);
        a[0] = f2bf(x0.x);  a[1] = f2bf(x0.y);  a[2]  = f2bf(x0.z);  a[3]  = f2bf(x0.w);
        a[4] = f2bf(x1.x);  a[5] = f2bf(x1.y);  a[6]  = f2bf(x1.z);  a[7]  = f2bf(x1.w);
        a[8] = f2bf(x2.x);  a[9] = f2bf(x2.y);  a[10] = f2bf(x2.z);  a[11] = f2bf(x2.w);
        a[12] = f2bf(x3.x); a[13] = f2bf(x3.y); a[14] = f2bf(x3.z); a[15] = f2bf(x3.w);
    } else {
        #pragma unroll
        for (int i = 0; i < 8; ++i) {
            const int k1 = k0 + hf * 16 + i;
            const int k2 = k0 + hf * 16 + 8 + i;
            float f1 = 0.f, f2 = 0.f;
            if (!GUARD || k1 < Kd) f1 = TA ? A[(long)k1 * lda + arow] : A[(long)arow * lda + k1];
            if (!GUARD || k2 < Kd) f2 = TA ? A[(long)k2 * lda + arow] : A[(long)arow * lda + k2];
            a[i]     = f2bf(f1);
            a[i + 8] = f2bf(f2);
        }
    }
    return a;
}

// =====================================================================
// WMMA GEMM: C[M,N] = post(alpha*op(A)@op(B) + betaBias*bias[col]
//                          + gammaAux*aux[row,col]), post = optional relu.
// Block = 128 threads = 4 waves stacked along M.
// WIDE (N%32==0, !TB): each wave owns a 16x32 output tile -> the A fragment
//   feeds TWO v_wmma ops (A reuse; halves A-side vmem+cvt per matrix op).
//   B tile (32xK-step x 32 cols f32) staged to LDS by the Tensor Data Mover
//   (wave0: tensor_load_to_lds + s_wait_tensorcnt 0), cooperative fallback.
// Narrow path: 16x16 tile, 32x16 LDS B tile, same staging.
// TB: per-lane k is contiguous -> direct b128 global loads, no LDS.
// =====================================================================
template<bool TA, bool TB, bool WIDE>
__global__ __launch_bounds__(128)
void wmma_gemm(const float* __restrict__ A, const float* __restrict__ B,
               const float* __restrict__ bias, const float* __restrict__ aux,
               float* __restrict__ C,
               int M, int Nd, int Kd, int lda, int ldb, int ldc,
               float alpha, float betaBias, float gammaAux, int relu)
{
    constexpr int LW = WIDE ? 32 : 16;           // B tile width (cols)
    __shared__ float ldsB[32 * LW];
    const int wave = threadIdx.x >> 5;
    const int lane = threadIdx.x & 31;
    const int tilesN = Nd / LW;
    const int mb = blockIdx.x / tilesN;
    const int tn = (blockIdx.x % tilesN) * LW;
    const int tmRaw = (mb << 6) + (wave << 4);
    const bool storeOK = tmRaw < M;
    const int tm = storeOK ? tmRaw : (M - 16);   // clamped waves duplicate work,
                                                 // keep barriers uniform, skip store
    const int hf = lane >> 4;
    const int l  = lane & 15;
    const int arow = tm + l;
    const int bcol = tn + l;

    v8f acc0 = {};
    v8f acc1 = {};
    const int k0f = Kd & ~31;                    // guard-free K span

    for (int k0 = 0; k0 < k0f; k0 += 32) {
        if (!TB) {
            __syncthreads();                     // prior reads of ldsB done
#if HAS_TDM
            if (wave == 0) {
                tdm_load_tile(B + (long)k0 * ldb + tn,
                              (unsigned)(uintptr_t)(void*)ldsB,
                              (unsigned)(Nd - tn), (long)Kd - k0, (unsigned)ldb,
                              (unsigned)LW);
                __builtin_amdgcn_s_wait_tensorcnt(0);
            }
#else
            for (int i = threadIdx.x; i < 32 * LW; i += 128) {
                const int kk = i / LW, nn = i % LW;
                ldsB[i] = B[(long)(k0 + kk) * ldb + tn + nn];
            }
#endif
            __syncthreads();
        }
        if (!TA && (k0 + 32) < Kd)
            __builtin_prefetch((const void*)(A + (long)arow * lda + k0 + 32), 0, 3);

        v16bf a = load_a_frag<TA, false>(A, lda, arow, k0, hf, Kd);
        if (TB) {
            v16bf b;
            const float* bp = B + (long)bcol * ldb + k0 + hf * 16;
            float4 y0 = *(const float4*)(bp);
            float4 y1 = *(const float4*)(bp + 4);
            float4 y2 = *(const float4*)(bp + 8);
            float4 y3 = *(const float4*)(bp + 12);
            b[0] = f2bf(y0.x);  b[1] = f2bf(y0.y);  b[2]  = f2bf(y0.z);  b[3]  = f2bf(y0.w);
            b[4] = f2bf(y1.x);  b[5] = f2bf(y1.y);  b[6]  = f2bf(y1.z);  b[7]  = f2bf(y1.w);
            b[8] = f2bf(y2.x);  b[9] = f2bf(y2.y);  b[10] = f2bf(y2.z);  b[11] = f2bf(y2.w);
            b[12] = f2bf(y3.x); b[13] = f2bf(y3.y); b[14] = f2bf(y3.z); b[15] = f2bf(y3.w);
            acc0 = __builtin_amdgcn_wmma_f32_16x16x32_bf16(false, a, false, b,
                                                           (short)0, acc0, false, false);
        } else {
            v16bf b0;
            #pragma unroll
            for (int j = 0; j < 16; ++j)
                b0[j] = f2bf(ldsB[(hf * 16 + j) * LW + l]);
            acc0 = __builtin_amdgcn_wmma_f32_16x16x32_bf16(false, a, false, b0,
                                                           (short)0, acc0, false, false);
            if (WIDE) {
                v16bf b1;
                #pragma unroll
                for (int j = 0; j < 16; ++j)
                    b1[j] = f2bf(ldsB[(hf * 16 + j) * LW + 16 + l]);
                acc1 = __builtin_amdgcn_wmma_f32_16x16x32_bf16(false, a, false, b1,
                                                               (short)0, acc1, false, false);
            }
        }
    }

    if (k0f < Kd) {                              // ragged-K tail (NFEAT=500 and NCLS=16)
        const int k0 = k0f;
        if (!TB) {
            __syncthreads();
#if HAS_TDM
            if (wave == 0) {
                tdm_load_tile(B + (long)k0 * ldb + tn,
                              (unsigned)(uintptr_t)(void*)ldsB,
                              (unsigned)(Nd - tn), (long)Kd - k0, (unsigned)ldb,
                              (unsigned)LW);
                __builtin_amdgcn_s_wait_tensorcnt(0);
            }
#else
            for (int i = threadIdx.x; i < 32 * LW; i += 128) {
                const int kk = i / LW, nn = i % LW;
                ldsB[i] = (k0 + kk < Kd) ? B[(long)(k0 + kk) * ldb + tn + nn] : 0.f;
            }
#endif
            __syncthreads();
        }
        v16bf a = load_a_frag<TA, true>(A, lda, arow, k0, hf, Kd);
        if (TB) {
            v16bf b;
            #pragma unroll
            for (int j = 0; j < 16; ++j) {
                const int k = k0 + hf * 16 + j;
                b[j] = f2bf((k < Kd) ? B[(long)bcol * ldb + k] : 0.f);
            }
            acc0 = __builtin_amdgcn_wmma_f32_16x16x32_bf16(false, a, false, b,
                                                           (short)0, acc0, false, false);
        } else {
            v16bf b0;
            #pragma unroll
            for (int j = 0; j < 16; ++j)
                b0[j] = f2bf(ldsB[(hf * 16 + j) * LW + l]);   // TDM zero-filled OOB
            acc0 = __builtin_amdgcn_wmma_f32_16x16x32_bf16(false, a, false, b0,
                                                           (short)0, acc0, false, false);
            if (WIDE) {
                v16bf b1;
                #pragma unroll
                for (int j = 0; j < 16; ++j)
                    b1[j] = f2bf(ldsB[(hf * 16 + j) * LW + 16 + l]);
                acc1 = __builtin_amdgcn_wmma_f32_16x16x32_bf16(false, a, false, b1,
                                                               (short)0, acc1, false, false);
            }
        }
    }

    if (!storeOK) return;
    const float bv0 = bias ? bias[bcol] : 0.f;
    const float bv1 = (WIDE && bias) ? bias[bcol + 16] : 0.f;
    #pragma unroll
    for (int v = 0; v < 8; ++v) {
        const int row = tm + hf * 8 + v;         // C layout: vgpr v -> M=v / 8+v
        float r = alpha * acc0[v] + betaBias * bv0;
        if (aux) r += gammaAux * aux[(long)row * ldc + bcol];
        if (relu) r = fmaxf(r, 0.f);
        C[(long)row * ldc + bcol] = r;
        if (WIDE) {
            float r1 = alpha * acc1[v] + betaBias * bv1;
            if (aux) r1 += gammaAux * aux[(long)row * ldc + bcol + 16];
            if (relu) r1 = fmaxf(r1, 0.f);
            C[(long)row * ldc + bcol + 16] = r1;
        }
    }
}

// ---------------- elementwise / edge kernels ----------------
__global__ void zero_f32(float* __restrict__ p, long n)
{
    long i = (long)blockIdx.x * blockDim.x + threadIdx.x;
    if (i < n) p[i] = 0.f;
}

__global__ void edge_conf(const int* __restrict__ ei, const float* __restrict__ lx,
                          const float* __restrict__ att_w, const float* __restrict__ conf_param,
                          const float* __restrict__ conf_in, float* __restrict__ conf)
{
    int e = blockIdx.x * blockDim.x + threadIdx.x;
    if (e >= EE) return;
    const int r = ei[e], c = ei[EE + e];
    float a = 0.f;
    #pragma unroll
    for (int j = 0; j < NCLS; ++j) {
        float t = lx[r * NCLS + j] + lx[c * NCLS + j];
        a += fmaxf(t, 0.f) * att_w[j];
    }
    const float sg = 1.f / (1.f + __expf(-a));
    const float e0 = __expf(conf_param[0]);
    const float e1 = __expf(conf_param[1]);
    const float e2 = __expf(conf_param[2]);
    const float inv = 1.f / (e0 + e1 + e2);
    conf[e] = sg * e0 * inv + conf_in[e] * e1 * inv + conf_in[EE + e] * e2 * inv;
}

__global__ void edge_score(const int* __restrict__ ei, const float* __restrict__ ew,
                           const float* __restrict__ P, const float* __restrict__ Q,
                           float* __restrict__ score, float* __restrict__ s2)
{
    int e = blockIdx.x * blockDim.x + threadIdx.x;
    if (e >= EE) return;
    const int r = ei[e], c = ei[EE + e];
    const float4* p = (const float4*)(P + (long)r * KK);
    const float4* q = (const float4*)(Q + (long)c * KK);
    float s = 0.f;
    #pragma unroll
    for (int k = 0; k < KK / 4; ++k) {
        float4 pv = p[k], qv = q[k];
        s += pv.x * qv.x + pv.y * qv.y + pv.z * qv.z + pv.w * qv.w;
    }
    const float sc = ew[e] - s;
    score[e] = sc;
    s2[e]    = sc * sc;
}

__global__ void edge_vals(const float* __restrict__ score, const float* __restrict__ s2,
                          const float* __restrict__ conf, const float* __restrict__ thr,
                          const float* __restrict__ b_param, float* __restrict__ vals)
{
    int e = blockIdx.x * blockDim.x + threadIdx.x;
    if (e >= EE) return;
    const float ce = (s2[e] < thr[0] * conf[e]) ? 1.f : 0.f;
    vals[e] = (score[e] + b_param[0]) * ce;
}

__global__ void scatter_edges(const int* __restrict__ ei, const float* __restrict__ coef,
                              const float* __restrict__ S, float* __restrict__ out, int rowDst)
{
    long idx = (long)blockIdx.x * blockDim.x + threadIdx.x;
    if (idx >= (long)EE * KK) return;
    const int e = (int)(idx >> 6);
    const int k = (int)(idx & 63);
    const int r = ei[e], c = ei[EE + e];
    const int dst = rowDst ? r : c;
    const int src = rowDst ? c : r;
    atomicAdd(&out[(long)dst * KK + k], coef[e] * S[(long)src * KK + k]);
}

// thr[0] = q3 + 1.5*(q3-q1) of s2[], 32-step value bisection over [0,max]
// (single resident WG; only 2 order stats needed, avoids full 256K sort)
__global__ __launch_bounds__(1024, 1)
void quartile_thr(const float* __restrict__ s2, float* __restrict__ thr)
{
    __shared__ float red[1024];
    __shared__ unsigned cnt;
    const int t = threadIdx.x, nt = blockDim.x;

    float m = 0.f;
    for (int i = t; i < EE; i += nt) m = fmaxf(m, s2[i]);
    red[t] = m; __syncthreads();
    for (int s = nt >> 1; s > 0; s >>= 1) {
        if (t < s) red[t] = fmaxf(red[t], red[t + s]);
        __syncthreads();
    }
    const float smax = red[0];
    __syncthreads();

    float q[2];
    const int ranks[2] = { EE / 4, 3 * (EE / 4) };
    for (int qi = 0; qi < 2; ++qi) {
        float lo = 0.f, hi = smax;
        for (int it = 0; it < 32; ++it) {
            const float mid = 0.5f * (lo + hi);
            if (t == 0) cnt = 0;
            __syncthreads();
            unsigned local = 0;
            for (int i = t; i < EE; i += nt) local += (s2[i] <= mid) ? 1u : 0u;
            atomicAdd(&cnt, local);
            __syncthreads();
            const unsigned c = cnt;
            __syncthreads();
            if ((int)c > ranks[qi]) hi = mid; else lo = mid;
        }
        q[qi] = hi;
    }
    if (t == 0) thr[0] = q[1] + 1.5f * (q[1] - q[0]);
}

// Out = inv(scale*A + diagAdd*I), 64x64 Gauss-Jordan in LDS (single WG)
__global__ __launch_bounds__(1024, 1)
void invert64(const float* __restrict__ A, float* __restrict__ Out,
              float scale, float diagAdd)
{
    __shared__ float Mx[64][129];
    __shared__ float fac[64];
    const int t = threadIdx.x, nt = blockDim.x;

    for (int idx = t; idx < 64 * 128; idx += nt) {
        const int r = idx >> 7, c = idx & 127;
        float v;
        if (c < 64) v = scale * A[r * 64 + c] + ((r == c) ? diagAdd : 0.f);
        else        v = ((c - 64) == r) ? 1.f : 0.f;
        Mx[r][c] = v;
    }
    __syncthreads();

    for (int p = 0; p < 64; ++p) {
        for (int r = t; r < 64; r += nt) fac[r] = Mx[r][p];
        __syncthreads();
        const float ipiv = 1.f / Mx[p][p];
        for (int c = t; c < 128; c += nt) Mx[p][c] *= ipiv;
        __syncthreads();
        for (int idx = t; idx < 64 * 128; idx += nt) {
            const int r = idx >> 7, c = idx & 127;
            if (r != p) Mx[r][c] -= fac[r] * Mx[p][c];
        }
        __syncthreads();
    }
    for (int idx = t; idx < 64 * 64; idx += nt) {
        const int r = idx >> 6, c = idx & 63;
        Out[r * 64 + c] = Mx[r][64 + c];
    }
}

// =====================================================================
extern "C" void kernel_launch(void* const* d_in, const int* in_sizes, int n_in,
                              void* d_out, int out_size, void* d_ws, size_t ws_size,
                              hipStream_t stream)
{
    const float* x          = (const float*)d_in[0];
    const float* edge_w     = (const float*)d_in[1];
    const float* adj_vals   = (const float*)d_in[2];
    const float* connect    = (const float*)d_in[3];
    const float* conf_in    = (const float*)d_in[4];
    const float* lin1_w     = (const float*)d_in[5];
    const float* lin1_b     = (const float*)d_in[6];
    const float* lin2_w     = (const float*)d_in[7];
    const float* lin2_b     = (const float*)d_in[8];
    const float* v_w        = (const float*)d_in[9];
    const float* v_b        = (const float*)d_in[10];
    const float* att_w      = (const float*)d_in[11];
    const float* left_w     = (const float*)d_in[12];
    const float* lin_a_w    = (const float*)d_in[13];
    const float* lin_a_b    = (const float*)d_in[14];
    const float* conf_param = (const float*)d_in[15];
    const float* b_param    = (const float*)d_in[16];
    const int*   edge_index = (const int*)d_in[17];
    float* out = (float*)d_out;

    float* ws = (float*)d_ws;
    long off = 0;
    auto take = [&](long n) { float* p = ws + off; off += n; return p; };
    float* xa    = take((long)NN * NHID);
    float* raw   = take((long)NN * NHID);
    float* xc    = take((long)NN * NCLS);
    float* lx    = take((long)NN * NCLS);
    float* conf  = take(EE);
    float* V     = take((long)NN * KK);
    float* U     = take((long)NN * KK);
    float* Wn    = take((long)NN * KK);
    float* temp  = take((long)NN * KK);
    float* score = take(EE);
    float* s2    = take(EE);
    float* vals  = take(EE);
    float* VtV   = take(64 * 64);
    float* inver = take(64 * 64);
    float* xcTV  = take(NCLS * KK);
    float* thr   = take(8);

    auto gemm = [&](const float* A, const float* B, const float* bias, const float* aux,
                    float* C, int M, int Nd, int Kd, int lda, int ldb, int ldc,
                    int tA, int tB, float alpha, float bb, float ga, int relu) {
        const bool wide = (!tB) && (Nd % 32 == 0);
        const int tilesN = Nd / (wide ? 32 : 16);
        const int mBlocks = (M + 63) / 64;
        dim3 g(mBlocks * tilesN), t(128);
        if (!tA && !tB && wide)
            wmma_gemm<false, false, true><<<g, t, 0, stream>>>(A, B, bias, aux, C, M, Nd, Kd,
                                                               lda, ldb, ldc, alpha, bb, ga, relu);
        else if (tA && !tB && wide)
            wmma_gemm<true, false, true><<<g, t, 0, stream>>>(A, B, bias, aux, C, M, Nd, Kd,
                                                              lda, ldb, ldc, alpha, bb, ga, relu);
        else if (!tA && !tB)
            wmma_gemm<false, false, false><<<g, t, 0, stream>>>(A, B, bias, aux, C, M, Nd, Kd,
                                                                lda, ldb, ldc, alpha, bb, ga, relu);
        else if (tA && !tB)
            wmma_gemm<true, false, false><<<g, t, 0, stream>>>(A, B, bias, aux, C, M, Nd, Kd,
                                                               lda, ldb, ldc, alpha, bb, ga, relu);
        else if (!tA)
            wmma_gemm<false, true, false><<<g, t, 0, stream>>>(A, B, bias, aux, C, M, Nd, Kd,
                                                               lda, ldb, ldc, alpha, bb, ga, relu);
        else
            wmma_gemm<true, true, false><<<g, t, 0, stream>>>(A, B, bias, aux, C, M, Nd, Kd,
                                                              lda, ldb, ldc, alpha, bb, ga, relu);
    };
    auto zero = [&](float* p, long n) {
        zero_f32<<<(int)((n + 255) / 256), 256, 0, stream>>>(p, n);
    };
    const int eBlocks = (EE + 255) / 256;
    const int sBlocks = (int)(((long)EE * KK + 255) / 256);

    // ---- front end ----
    gemm(connect, lin_a_w, lin_a_b, nullptr, xa, NN, NHID, NN, NN, NHID, NHID, 0, 0, 1.f, 1.f, 0.f, 0);
    gemm(x, lin1_w, lin1_b, xa, raw, NN, NHID, NFEAT, NFEAT, NHID, NHID, 0, 0,
         1.f - LAMDAC, 1.f - LAMDAC, LAMDAC, 1);
    gemm(raw, lin2_w, lin2_b, nullptr, xc, NN, NCLS, NHID, NHID, NCLS, NCLS, 0, 0, 1.f, 1.f, 0.f, 0);
    gemm(xc, left_w, nullptr, nullptr, lx, NN, NCLS, NCLS, NCLS, NCLS, NCLS, 0, 0, 1.f, 0.f, 0.f, 0);
    edge_conf<<<eBlocks, 256, 0, stream>>>(edge_index, lx, att_w, conf_param, conf_in, conf);
    gemm(raw, v_w, v_b, nullptr, V, NN, KK, NHID, NHID, KK, KK, 0, 0, 1.f, 1.f, 0.f, 0);

    // ---- U init: U = segsum(adj*V[col] by row) @ inv(V^T V) ----
    zero(temp, (long)NN * KK);
    scatter_edges<<<sBlocks, 256, 0, stream>>>(edge_index, adj_vals, V, temp, 1);
    gemm(V, V, nullptr, nullptr, VtV, 64, 64, NN, KK, KK, 64, 1, 0, 1.f, 0.f, 0.f, 0);
    invert64<<<1, 1024, 0, stream>>>(VtV, inver, 1.f, 0.f);
    gemm(temp, inver, nullptr, nullptr, U, NN, KK, KK, KK, KK, KK, 0, 0, 1.f, 0.f, 0.f, 0);

    // ---- GAMMA iterations ----
    for (int j = 0; j < 4; ++j) {
        if ((j & 1) == 0) {                       // update U
            edge_score<<<eBlocks, 256, 0, stream>>>(edge_index, edge_w, U, V, score, s2);
            quartile_thr<<<1, 1024, 0, stream>>>(s2, thr);
            edge_vals<<<eBlocks, 256, 0, stream>>>(score, s2, conf, thr, b_param, vals);
            zero(temp, (long)NN * KK);
            scatter_edges<<<sBlocks, 256, 0, stream>>>(edge_index, vals, V, temp, 1);
            gemm(V, V, nullptr, nullptr, VtV, 64, 64, NN, KK, KK, 64, 1, 0, 1.f, 0.f, 0.f, 0);
            gemm(xc, V, nullptr, nullptr, xcTV, NCLS, KK, NN, NCLS, KK, KK, 1, 0, 1.f, 0.f, 0.f, 0);
            gemm(U, VtV, nullptr, temp, Wn, NN, KK, KK, KK, 64, KK, 0, 0, 1.f, 0.f, 1.f, 0);
            gemm(xc, xcTV, nullptr, Wn, Wn, NN, KK, NCLS, NCLS, KK, KK, 0, 0, EPSC, 0.f, 1.f, 0);
            invert64<<<1, 1024, 0, stream>>>(VtV, inver, 1.f + EPSC, BETAC);
            gemm(Wn, inver, nullptr, nullptr, U, NN, KK, KK, KK, KK, KK, 0, 0, 1.f, 0.f, 0.f, 0);
        } else {                                  // update V
            gemm(U, U, nullptr, nullptr, VtV, 64, 64, NN, KK, KK, 64, 1, 0, 1.f, 0.f, 0.f, 0);
            invert64<<<1, 1024, 0, stream>>>(VtV, inver, 1.f + EPSC, BETAC);
            edge_score<<<eBlocks, 256, 0, stream>>>(edge_index, edge_w, V, U, score, s2);
            quartile_thr<<<1, 1024, 0, stream>>>(s2, thr);
            edge_vals<<<eBlocks, 256, 0, stream>>>(score, s2, conf, thr, b_param, vals);
            zero(temp, (long)NN * KK);
            scatter_edges<<<sBlocks, 256, 0, stream>>>(edge_index, vals, U, temp, 0);
            gemm(xc, U, nullptr, nullptr, xcTV, NCLS, KK, NN, NCLS, KK, KK, 1, 0, 1.f, 0.f, 0.f, 0);
            gemm(V, VtV, nullptr, temp, Wn, NN, KK, KK, KK, 64, KK, 0, 0, 1.f, 0.f, 1.f, 0);
            gemm(xc, xcTV, nullptr, Wn, Wn, NN, KK, NCLS, NCLS, KK, KK, 0, 0, EPSC, 0.f, 1.f, 0);
            gemm(Wn, inver, nullptr, nullptr, V, NN, KK, KK, KK, KK, KK, 0, 0, 1.f, 0.f, 0.f, 0);
        }
    }

    // ---- output: out = 0.9 * U @ (xc^T V)^T + 0.1 * xc ----
    gemm(xc, V, nullptr, nullptr, xcTV, NCLS, KK, NN, NCLS, KK, KK, 1, 0, 1.f, 0.f, 0.f, 0);
    gemm(U, xcTV, nullptr, xc, out, NN, NCLS, KK, KK, KK, NCLS, 0, 1,
         1.f - ALPHAC, 0.f, ALPHAC, 0);
}
